// Edge_Transformer_82248623718559
// MI455X (gfx1250) — matrix operand order repaired and checked by reading, not compiled, not run on hardware
//
#include <hip/hip_runtime.h>

// ---------------------------------------------------------------------------
// Edge-MLP on gfx1250 with bf16 WMMA (V_WMMA_F32_16X16X32_BF16).
//   feats[E,80] = [nodes[snd] | nodes[rcv] | edges]   (K padded to 96)
//   h   = sigmoid(feats @ W1 + b1)   [E,160]
//   out = h @ W2 + b2                [E,32]
//
// prep kernel: swizzle W1/W2 -> bf16 WMMA B-fragment layout once into d_ws.
// main kernel: 128 threads (4 waves), loops over 4 tiles of 64 edges.
//   - weights: one contiguous 40KB global->LDS copy via async-to-LDS
//   - feats:   gathered 8 floats at a time -> packed bf16 -> ds_store_b128
//   - GEMM1:   10 N-tiles x 3 K-steps WMMA, bias-init accumulators
//   - sigmoid epilogue re-swizzled (wave-private) into A-fragment layout
//   - GEMM2:   2 N-tiles x 5 K-steps WMMA, direct f32 stores
// ---------------------------------------------------------------------------

#define D_NODE 32
#define D_EDGE 16
#define D_IN   80
#define D_INP  96
#define D_HID  160
#define D_OUTF 32

#define WAVES  4
#define BE     (WAVES * 16)    // 64 edges per tile
#define NTHR   (WAVES * 32)    // 128 threads
#define TILES  4               // edge tiles per block

#define W1F_ELEMS (3 * 10 * 512)            // 15360 bf16
#define W2F_ELEMS (5 * 2 * 512)             // 5120 bf16
#define WF_ELEMS  (W1F_ELEMS + W2F_ELEMS)   // 20480 bf16
#define WF_BYTES  (WF_ELEMS * 2)            // 40960 B

typedef __attribute__((ext_vector_type(16))) __bf16 v16bf;
typedef __attribute__((ext_vector_type(8)))  __bf16 v8bf;
typedef __attribute__((ext_vector_type(8)))  float  v8f;
typedef __attribute__((ext_vector_type(4)))  int    v4i;

typedef __attribute__((address_space(1))) v4i* gv4i_p;   // global int4*
typedef __attribute__((address_space(3))) v4i* lv4i_p;   // LDS int4*

#if defined(__has_builtin)
#if __has_builtin(__builtin_amdgcn_global_load_async_to_lds_b128)
#define HAVE_ASYNC_LDS 1
#endif
#endif

// ---------------------------------------------------------------------------
// Prep: swizzle weights into WMMA B-fragment-major bf16, once, into d_ws.
// B-frag layout used here (contiguous k) for 32x16 tile: lane = (k%32>=16)*16 + n%16, i = k%16.
// ---------------------------------------------------------------------------
__global__ __launch_bounds__(256) void prep_weights_kernel(
    const float* __restrict__ W1, const float* __restrict__ W2,
    __bf16* __restrict__ wf)
{
  int t = blockIdx.x * 256 + threadIdx.x;
  if (t < D_INP * D_HID) {
    int k = t / D_HID, n = t - k * D_HID;
    float v = (k < D_IN) ? W1[k * D_HID + n] : 0.0f;   // zero-pad K 80..95
    int kt = k >> 5, kk = k & 31, nt = n >> 4, nn = n & 15;
    int lane = ((kk >> 4) << 4) | nn;
    int i    = kk & 15;
    wf[((kt * 10 + nt) << 9) + (lane << 4) + i] = (__bf16)v;
  } else if (t < D_INP * D_HID + D_HID * D_OUTF) {
    int idx = t - D_INP * D_HID;
    int k = idx >> 5, n = idx & 31;
    float v = W2[idx];
    int kt = k >> 5, kk = k & 31, nt = n >> 4, nn = n & 15;
    int lane = ((kk >> 4) << 4) | nn;
    int i    = kk & 15;
    wf[W1F_ELEMS + ((kt * 2 + nt) << 9) + (lane << 4) + i] = (__bf16)v;
  }
}

// ---------------------------------------------------------------------------
// Main kernel. FROMWS: stage swizzled weights from d_ws (fast path); else
// swizzle from raw W1/W2 in-block (fallback if ws too small).
// ---------------------------------------------------------------------------
template <bool FROMWS>
__global__ __launch_bounds__(NTHR) void edge_mlp_wmma_kernel(
    const float* __restrict__ nodes,
    const float* __restrict__ edges,
    const int*   __restrict__ senders,
    const int*   __restrict__ receivers,
    const __bf16* __restrict__ wfg,     // swizzled weights in d_ws
    const float* __restrict__ W1,
    const float* __restrict__ W2,
    const float* __restrict__ b1,
    const float* __restrict__ b2,
    float* __restrict__ out,
    int nEdges)
{
  __shared__ __attribute__((aligned(32))) __bf16 wf[WF_ELEMS];        // 40960 B
  __shared__ __attribute__((aligned(32))) __bf16 ah[WAVES * 5 * 512]; // 20480 B
  __shared__ float b1s[D_HID];
  __shared__ float b2s[D_OUTF];
  __shared__ int   sid[BE];
  __shared__ int   rid[BE];

  const int tid = threadIdx.x;

  // ---- one-time staging: weights + biases ---------------------------------
  if (FROMWS) {
#if defined(HAVE_ASYNC_LDS)
    for (int i = tid * 16; i < WF_BYTES; i += NTHR * 16) {
      __builtin_amdgcn_global_load_async_to_lds_b128(
          (gv4i_p)((char*)wfg + i),
          (lv4i_p)((char*)wf + i),
          0, 0);
    }
#if defined(__has_builtin) && __has_builtin(__builtin_amdgcn_s_wait_asynccnt)
    __builtin_amdgcn_s_wait_asynccnt(0);
#else
    asm volatile("s_wait_asynccnt 0" ::: "memory");
#endif
#else
    const uint4* src = (const uint4*)wfg;
    uint4*       dst = (uint4*)wf;
    for (int i = tid; i < WF_BYTES / 16; i += NTHR) dst[i] = src[i];
#endif
  } else {
    for (int idx = tid; idx < D_INP * D_HID; idx += NTHR) {
      int k = idx / D_HID, n = idx - k * D_HID;
      float v = (k < D_IN) ? W1[k * D_HID + n] : 0.0f;
      int kt = k >> 5, kk = k & 31, nt = n >> 4, nn = n & 15;
      int lane = ((kk >> 4) << 4) | nn;
      wf[((kt * 10 + nt) << 9) + (lane << 4) + (kk & 15)] = (__bf16)v;
    }
    for (int idx = tid; idx < D_HID * D_OUTF; idx += NTHR) {
      int k = idx >> 5, n = idx & 31;
      float v = W2[idx];
      int kt = k >> 5, kk = k & 31, nt = n >> 4, nn = n & 15;
      int lane = ((kk >> 4) << 4) | nn;
      wf[W1F_ELEMS + ((kt * 2 + nt) << 9) + (lane << 4) + (kk & 15)] = (__bf16)v;
    }
  }
  for (int j = tid; j < D_HID; j += NTHR) b1s[j] = b1[j];
  if (tid < D_OUTF) b2s[tid] = b2[tid];

  const int wave = tid >> 5;
  const int lane = tid & 31;
  const int nlo  = lane & 15;          // C/D column within tile
  const int mhi  = (lane >> 4) << 3;   // C/D row offset for hi half-wave

  // ---- loop over edge tiles ----------------------------------------------
  for (int t = 0; t < TILES; ++t) {
    const int e0 = (blockIdx.x * TILES + t) * BE;

    for (int m = tid; m < BE; m += NTHR) {
      int e = e0 + m; if (e >= nEdges) e = nEdges - 1;   // clamp, EXEC stays full
      sid[m] = senders[e];
      rid[m] = receivers[e];
    }
    __syncthreads();   // (a) indices ready; previous tile's ah reads done

    // -- gather: 8-float groups -> packed bf16 -> one ds_store_b128 each ----
    // group g of edge m covers k = 8g..8g+7; contiguous 16B in A-frag layout.
    for (int idx = tid; idx < BE * 12; idx += NTHR) {
      int m = idx / 12, g = idx - m * 12;
      int f0 = g << 3;
      v8bf p;
      if (g < 10) {
        const float* src;
        if (g < 4) {
          src = nodes + (long long)sid[m] * D_NODE + f0;
        } else if (g < 8) {
          src = nodes + (long long)rid[m] * D_NODE + (f0 - 32);
        } else {
          int e = e0 + m; if (e >= nEdges) e = nEdges - 1;
          src = edges + (long long)e * D_EDGE + (f0 - 64);
        }
        float4 u0 = ((const float4*)src)[0];
        float4 u1 = ((const float4*)src)[1];
        p[0] = (__bf16)u0.x; p[1] = (__bf16)u0.y;
        p[2] = (__bf16)u0.z; p[3] = (__bf16)u0.w;
        p[4] = (__bf16)u1.x; p[5] = (__bf16)u1.y;
        p[6] = (__bf16)u1.z; p[7] = (__bf16)u1.w;
      } else {
#pragma unroll
        for (int j = 0; j < 8; ++j) p[j] = (__bf16)0.0f;  // K padding 80..95
      }
      int mt  = m >> 4;
      int kt  = f0 >> 5, kk0 = f0 & 31;
      int hi  = (kk0 >> 4) & 1;
      int i0  = ((kk0 >> 3) & 1) << 3;    // 0 or 8
      int ln  = (hi << 4) | (m & 15);
      *(v8bf*)&ah[((mt * 5 + kt) << 9) + (ln << 4) + i0] = p;
    }
    __syncthreads();   // (b) feats staged

    // -- GEMM1: [16x96] x [96x160] ------------------------------------------
    v16bf a0 = *(const v16bf*)&ah[((wave * 5 + 0) << 9) + (lane << 4)];
    v16bf a1 = *(const v16bf*)&ah[((wave * 5 + 1) << 9) + (lane << 4)];
    v16bf a2 = *(const v16bf*)&ah[((wave * 5 + 2) << 9) + (lane << 4)];

    v8f acc[10];
#pragma unroll
    for (int nt = 0; nt < 10; ++nt) {
      float bb = b1s[nt * 16 + nlo];
      v8f c;
#pragma unroll
      for (int j = 0; j < 8; ++j) c[j] = bb;
      v16bf bw0 = *(const v16bf*)&wf[((0 * 10 + nt) << 9) + (lane << 4)];
      v16bf bw1 = *(const v16bf*)&wf[((1 * 10 + nt) << 9) + (lane << 4)];
      v16bf bw2 = *(const v16bf*)&wf[((2 * 10 + nt) << 9) + (lane << 4)];
      c = __builtin_amdgcn_wmma_f32_16x16x32_bf16(false, a0, false, bw0, (short)0, c, false, false);
      c = __builtin_amdgcn_wmma_f32_16x16x32_bf16(false, a1, false, bw1, (short)0, c, false, false);
      c = __builtin_amdgcn_wmma_f32_16x16x32_bf16(false, a2, false, bw2, (short)0, c, false, false);
      acc[nt] = c;
    }

    // -- sigmoid + re-swizzle hidden (wave-private region of ah) ------------
#pragma unroll
    for (int nt = 0; nt < 10; ++nt) {
#pragma unroll
      for (int v = 0; v < 8; ++v) {
        float h = 1.0f / (1.0f + __expf(-acc[nt][v]));
        int m   = v + mhi;
        int k2  = nt * 16 + nlo;
        int kt2 = k2 >> 5, kk2 = k2 & 31;
        int hi2 = (kk2 >> 4) & 1;
        int i2  = kk2 & 15;
        int ln2 = (hi2 << 4) | m;
        ah[((wave * 5 + kt2) << 9) + (ln2 << 4) + i2] = (__bf16)h;
      }
    }
    __syncthreads();   // (c) fence before h-fragment loads

    // -- GEMM2: [16x160] x [160x32] -----------------------------------------
    v16bf h0 = *(const v16bf*)&ah[((wave * 5 + 0) << 9) + (lane << 4)];
    v16bf h1 = *(const v16bf*)&ah[((wave * 5 + 1) << 9) + (lane << 4)];
    v16bf h2 = *(const v16bf*)&ah[((wave * 5 + 2) << 9) + (lane << 4)];
    v16bf h3 = *(const v16bf*)&ah[((wave * 5 + 3) << 9) + (lane << 4)];
    v16bf h4 = *(const v16bf*)&ah[((wave * 5 + 4) << 9) + (lane << 4)];

#pragma unroll
    for (int nt2 = 0; nt2 < 2; ++nt2) {
      float bb = b2s[nt2 * 16 + nlo];
      v8f c;
#pragma unroll
      for (int j = 0; j < 8; ++j) c[j] = bb;
      v16bf q0 = *(const v16bf*)&wf[W1F_ELEMS + ((0 * 2 + nt2) << 9) + (lane << 4)];
      v16bf q1 = *(const v16bf*)&wf[W1F_ELEMS + ((1 * 2 + nt2) << 9) + (lane << 4)];
      v16bf q2 = *(const v16bf*)&wf[W1F_ELEMS + ((2 * 2 + nt2) << 9) + (lane << 4)];
      v16bf q3 = *(const v16bf*)&wf[W1F_ELEMS + ((3 * 2 + nt2) << 9) + (lane << 4)];
      v16bf q4 = *(const v16bf*)&wf[W1F_ELEMS + ((4 * 2 + nt2) << 9) + (lane << 4)];
      c = __builtin_amdgcn_wmma_f32_16x16x32_bf16(false, h0, false, q0, (short)0, c, false, false);
      c = __builtin_amdgcn_wmma_f32_16x16x32_bf16(false, h1, false, q1, (short)0, c, false, false);
      c = __builtin_amdgcn_wmma_f32_16x16x32_bf16(false, h2, false, q2, (short)0, c, false, false);
      c = __builtin_amdgcn_wmma_f32_16x16x32_bf16(false, h3, false, q3, (short)0, c, false, false);
      c = __builtin_amdgcn_wmma_f32_16x16x32_bf16(false, h4, false, q4, (short)0, c, false, false);

#pragma unroll
      for (int v = 0; v < 8; ++v) {
        int m = v + mhi;
        long long e = (long long)(e0 + wave * 16 + m);
        if (e < nEdges)
          out[e * D_OUTF + nt2 * 16 + nlo] = c[v];
      }
    }
    // loop back: (a) barrier of next tile protects ah/sid reuse
  }
}

extern "C" void kernel_launch(void* const* d_in, const int* in_sizes, int n_in,
                              void* d_out, int out_size, void* d_ws, size_t ws_size,
                              hipStream_t stream) {
  const float* nodes     = (const float*)d_in[0];
  const float* edges     = (const float*)d_in[1];
  const int*   senders   = (const int*)d_in[2];
  const int*   receivers = (const int*)d_in[3];
  const float* W1        = (const float*)d_in[4];
  const float* b1        = (const float*)d_in[5];
  const float* W2        = (const float*)d_in[6];
  const float* b2        = (const float*)d_in[7];
  float* out = (float*)d_out;

  const int nEdges = in_sizes[2];
  const int blocks = (nEdges + TILES * BE - 1) / (TILES * BE);   // 6250

  if (ws_size >= (size_t)WF_BYTES) {
    __bf16* wf = (__bf16*)d_ws;
    prep_weights_kernel<<<(WF_ELEMS + 255) / 256, 256, 0, stream>>>(W1, W2, wf);
    edge_mlp_wmma_kernel<true><<<blocks, NTHR, 0, stream>>>(
        nodes, edges, senders, receivers, wf, W1, W2, b1, b2, out, nEdges);
  } else {
    edge_mlp_wmma_kernel<false><<<blocks, NTHR, 0, stream>>>(
        nodes, edges, senders, receivers, (const __bf16*)nullptr, W1, W2, b1, b2, out, nEdges);
  }
}
